// MPGraph_19104014532840
// MI455X (gfx1250) — matrix operand order repaired and checked by reading, not compiled, not run on hardware
//
#include <hip/hip_runtime.h>

#define N_NODES 50000
#define N_EDGES 1600000

typedef __attribute__((ext_vector_type(16))) _Float16 v16h;
typedef __attribute__((ext_vector_type(8)))  _Float16 v8h;
typedef __attribute__((ext_vector_type(8)))  float    v8f;

union V16U { v16h v; v8h h[2]; };

__device__ __forceinline__ float lrelu_f(float v) { return v > 0.f ? v : 0.01f * v; }

// ---------------------------------------------------------------------------
// Repack edge-MLP weights into WMMA B-fragment order (f16).
// B (32x16, KxN) layout: lane l holds column n = l&15; halves h=0..15 hold
// K = (l>=16 ? 16 : 0) + h within the 32-K chunk. Fragment = 512 halves,
// lane-contiguous so the edge kernel loads it with one 32B LDS read per lane.
// w0f: 5 K-chunks x 5 N-tiles (160x80).  w1f: 3 K-chunks x 1 N-tile (96x16,
// rows 80..95 zero-padded).
// ---------------------------------------------------------------------------
__global__ void prep_weights_kernel(const float* __restrict__ W0,
                                    const float* __restrict__ W1,
                                    _Float16* __restrict__ w0f,
                                    _Float16* __restrict__ w1f) {
  int idx = blockIdx.x * 256 + threadIdx.x;
  const int total = 25 * 512 + 3 * 512;
  for (; idx < total; idx += gridDim.x * 256) {
    if (idx < 25 * 512) {
      int f = idx >> 9, rem = idx & 511;
      int l = rem >> 4, h = rem & 15;
      int kc = f / 5, nt = f % 5;
      int K = kc * 32 + ((l >= 16) ? 16 : 0) + h;
      int col = nt * 16 + (l & 15);
      w0f[idx] = (_Float16)W0[K * 80 + col];
    } else {
      int j = idx - 25 * 512;
      int f = j >> 9, rem = j & 511;
      int l = rem >> 4, h = rem & 15;
      int K = f * 32 + ((l >= 16) ? 16 : 0) + h;
      int col = l & 15;
      w1f[j] = (_Float16)((K < 80) ? W1[K * 16 + col] : 0.f);
    }
  }
}

// nc[n] = [ f16(lrelu(x[n])) | f16(lrelu(x[n])) ]  (n0 | n, both init to n0)
__global__ __launch_bounds__(256)
void node_init_kernel(const float* __restrict__ x, _Float16* __restrict__ nc) {
  const int n = blockIdx.x * 256 + threadIdx.x;
  if (n >= N_NODES) return;
  _Float16* d = nc + (size_t)n * 64;
  #pragma unroll
  for (int j = 0; j < 32; ++j) {
    float v = lrelu_f(x[(size_t)n * 32 + j]);
    d[j] = (_Float16)v;
    d[32 + j] = (_Float16)v;
  }
}

// Encoder edge MLP: 6 -> 18 -> 18 -> 16, lrelu after each layer. Scalar (tiny).
__global__ __launch_bounds__(256)
void encoder_kernel(const float* __restrict__ ea,
                    const float* __restrict__ W0, const float* __restrict__ b0,
                    const float* __restrict__ W1, const float* __restrict__ b1,
                    const float* __restrict__ W2, const float* __restrict__ b2,
                    _Float16* __restrict__ e0out) {
  __shared__ float sW0[6 * 18], sb0[18], sW1[18 * 18], sb1[18], sW2[18 * 16], sb2[16];
  const int tid = threadIdx.x;
  for (int i = tid; i < 108; i += 256) sW0[i] = W0[i];
  for (int i = tid; i < 324; i += 256) sW1[i] = W1[i];
  for (int i = tid; i < 288; i += 256) sW2[i] = W2[i];
  if (tid < 18) { sb0[tid] = b0[tid]; sb1[tid] = b1[tid]; }
  if (tid < 16) sb2[tid] = b2[tid];
  __syncthreads();
  const int e = blockIdx.x * 256 + tid;
  if (e >= N_EDGES) return;
  float in[6];
  #pragma unroll
  for (int j = 0; j < 6; ++j) in[j] = ea[(size_t)e * 6 + j];
  float h0[18];
  #pragma unroll
  for (int k = 0; k < 18; ++k) h0[k] = sb0[k];
  #pragma unroll
  for (int j = 0; j < 6; ++j) {
    const float vj = in[j];
    #pragma unroll
    for (int k = 0; k < 18; ++k) h0[k] = fmaf(vj, sW0[j * 18 + k], h0[k]);
  }
  #pragma unroll
  for (int k = 0; k < 18; ++k) h0[k] = lrelu_f(h0[k]);
  float h1[18];
  #pragma unroll
  for (int k = 0; k < 18; ++k) h1[k] = sb1[k];
  for (int j = 0; j < 18; ++j) {
    const float vj = h0[j];
    #pragma unroll
    for (int k = 0; k < 18; ++k) h1[k] = fmaf(vj, sW1[j * 18 + k], h1[k]);
  }
  #pragma unroll
  for (int k = 0; k < 18; ++k) h1[k] = lrelu_f(h1[k]);
  float o[16];
  #pragma unroll
  for (int m = 0; m < 16; ++m) o[m] = sb2[m];
  for (int j = 0; j < 18; ++j) {
    const float vj = h1[j];
    #pragma unroll
    for (int m = 0; m < 16; ++m) o[m] = fmaf(vj, sW2[j * 16 + m], o[m]);
  }
  _Float16* d = e0out + (size_t)e * 16;
  #pragma unroll
  for (int m = 0; m < 16; ++m) d[m] = (_Float16)lrelu_f(o[m]);
}

__global__ void zero_f32_kernel(float* __restrict__ p, int n) {
  const int i = blockIdx.x * 256 + threadIdx.x;
  if (i < n) p[i] = 0.f;
}

__global__ void count_kernel(const int* __restrict__ rowp, float* __restrict__ cnt) {
  const int e = blockIdx.x * 256 + threadIdx.x;
  if (e < N_EDGES) unsafeAtomicAdd(&cnt[rowp[e]], 1.f);
}

__global__ void inv_kernel(float* __restrict__ c) {
  const int n = blockIdx.x * 256 + threadIdx.x;
  if (n < N_NODES) c[n] = 1.f / fmaxf(c[n], 1.f);
}

// ---------------------------------------------------------------------------
// Edge update: eh=[nc[row] | nc[col] | e0 | e_prev] (160) -> 80 (lrelu) -> 16
// (lrelu), via v_wmma_f32_16x16x32_f16. 4 waves/block, 16 edges per wave.
// Fused epilogue: write e_cur (f16), output (f32, steps>=2), scatter-add to
// per-node accumulator.
// ---------------------------------------------------------------------------
__global__ __launch_bounds__(128)
void edge_step_kernel(const _Float16* __restrict__ nc,
                      const _Float16* __restrict__ e0,
                      const _Float16* __restrict__ eprev,
                      const int* __restrict__ rowp,
                      const int* __restrict__ colp,
                      const _Float16* __restrict__ w0f,
                      const _Float16* __restrict__ w1f,
                      const float* __restrict__ b0,
                      const float* __restrict__ b1,
                      _Float16* __restrict__ ecur,
                      float* __restrict__ outp,
                      float* __restrict__ sacc) {
  __shared__ __align__(32) _Float16 sh_eh[64][168];   // 64 edges x 160 (pad 168)
  __shared__ __align__(32) _Float16 sh_w0[25 * 512];  // B fragments GEMM1
  __shared__ __align__(32) _Float16 sh_w1[3 * 512];   // B fragments GEMM2
  __shared__ __align__(32) _Float16 sh_h[4][16 * 96]; // per-wave intermediate
  __shared__ float sh_b0[80];
  __shared__ float sh_b1[16];

  const int tid = threadIdx.x;
  for (int i = tid; i < 1600; i += 128) ((uint4*)sh_w0)[i] = ((const uint4*)w0f)[i];
  for (int i = tid; i < 192; i += 128)  ((uint4*)sh_w1)[i] = ((const uint4*)w1f)[i];
  if (tid < 80) sh_b0[tid] = b0[tid];
  if (tid < 16) sh_b1[tid] = b1[tid];

  const int ebase = blockIdx.x * 64;
  {
    const int le = tid >> 1, part = tid & 1;
    const int edge = ebase + le;
    if (part == 0) {
      const uint4* src = (const uint4*)(nc + (size_t)rowp[edge] * 64);
      uint4* dst = (uint4*)&sh_eh[le][0];
      #pragma unroll
      for (int i = 0; i < 8; ++i) dst[i] = src[i];
    } else {
      const uint4* src = (const uint4*)(nc + (size_t)colp[edge] * 64);
      uint4* dst = (uint4*)&sh_eh[le][64];
      #pragma unroll
      for (int i = 0; i < 8; ++i) dst[i] = src[i];
      const uint4* s0 = (const uint4*)(e0 + (size_t)edge * 16);
      uint4* d0 = (uint4*)&sh_eh[le][128];
      d0[0] = s0[0]; d0[1] = s0[1];
      const uint4* s1 = (const uint4*)(eprev + (size_t)edge * 16);
      uint4* d1 = (uint4*)&sh_eh[le][144];
      d1[0] = s1[0]; d1[1] = s1[1];
    }
  }
  __syncthreads();

  const int lane = tid & 31;
  const int wave = tid >> 5;
  const int r16  = lane & 15;
  const int hi   = lane >> 4;

  // A fragments: contiguous k order, same as the B packing — lane<16 holds K 0-15; lane>=16 K 16-31.
  v16h a[5];
  {
    const _Float16* arow = &sh_eh[wave * 16 + r16][0];
    #pragma unroll
    for (int kc = 0; kc < 5; ++kc) {
      V16U u;
      u.h[0] = *(const v8h*)(arow + kc * 32 + hi * 16);
      u.h[1] = *(const v8h*)(arow + kc * 32 + hi * 16 + 8);
      a[kc] = u.v;
    }
  }

  // GEMM1: 160 -> 80, bias in C, lrelu -> sh_h (K-padded to 96)
  #pragma unroll
  for (int nt = 0; nt < 5; ++nt) {
    const float bv = sh_b0[nt * 16 + r16];
    v8f c = {bv, bv, bv, bv, bv, bv, bv, bv};
    #pragma unroll
    for (int kc = 0; kc < 5; ++kc) {
      v16h b = *(const v16h*)(sh_w0 + (kc * 5 + nt) * 512 + lane * 16);
      c = __builtin_amdgcn_wmma_f32_16x16x32_f16(false, a[kc], false, b,
                                                 (short)0, c, false, false);
    }
    #pragma unroll
    for (int r = 0; r < 8; ++r)
      sh_h[wave][(hi * 8 + r) * 96 + nt * 16 + r16] = (_Float16)lrelu_f(c[r]);
  }
  #pragma unroll
  for (int r = 0; r < 8; ++r)
    sh_h[wave][(hi * 8 + r) * 96 + 80 + r16] = (_Float16)0.f;
  asm volatile("s_wait_dscnt 0" ::: "memory");  // intra-wave LDS RAW across lanes

  // GEMM2: 96 -> 16
  v16h a2[3];
  {
    const _Float16* hrow = &sh_h[wave][r16 * 96];
    #pragma unroll
    for (int kc = 0; kc < 3; ++kc) {
      V16U u;
      u.h[0] = *(const v8h*)(hrow + kc * 32 + hi * 16);
      u.h[1] = *(const v8h*)(hrow + kc * 32 + hi * 16 + 8);
      a2[kc] = u.v;
    }
  }
  const float bv1 = sh_b1[r16];
  v8f c2 = {bv1, bv1, bv1, bv1, bv1, bv1, bv1, bv1};
  #pragma unroll
  for (int kc = 0; kc < 3; ++kc) {
    v16h b = *(const v16h*)(sh_w1 + kc * 512 + lane * 16);
    c2 = __builtin_amdgcn_wmma_f32_16x16x32_f16(false, a2[kc], false, b,
                                                (short)0, c2, false, false);
  }

  // Epilogue: D layout — lane col = lane&15, VGPR r -> row hi*8 + r.
  const int tb = ebase + wave * 16 + hi * 8;
  #pragma unroll
  for (int r = 0; r < 8; ++r) {
    const int edge = tb + r;
    const float v = lrelu_f(c2[r]);
    ecur[(size_t)edge * 16 + r16] = (_Float16)v;
    if (outp) outp[(size_t)edge * 16 + r16] = v;
    unsafeAtomicAdd(&sacc[(size_t)rowp[edge] * 16 + r16], v);
  }
}

// Node update: nh=[nc | mean_msg] (80) -> 56 (relu) -> 32 (relu); scalar.
__global__ __launch_bounds__(256)
void node_step_kernel(_Float16* __restrict__ nc,
                      const float* __restrict__ sacc,
                      const float* __restrict__ invc,
                      const float* __restrict__ W0, const float* __restrict__ b0,
                      const float* __restrict__ W1, const float* __restrict__ b1) {
  __shared__ float sW0[80 * 56];
  __shared__ float sW1[56 * 32];
  __shared__ float sb0[56];
  __shared__ float sb1[32];
  const int tid = threadIdx.x;
  for (int i = tid; i < 80 * 56; i += 256) sW0[i] = W0[i];
  for (int i = tid; i < 56 * 32; i += 256) sW1[i] = W1[i];
  if (tid < 56) sb0[tid] = b0[tid];
  if (tid < 32) sb1[tid] = b1[tid];
  __syncthreads();
  const int n = blockIdx.x * 256 + tid;
  if (n >= N_NODES) return;
  float h[56];
  #pragma unroll
  for (int k = 0; k < 56; ++k) h[k] = sb0[k];
  const _Float16* ncr = nc + (size_t)n * 64;
  for (int j = 0; j < 64; ++j) {
    const float vj = (float)ncr[j];
    #pragma unroll
    for (int k = 0; k < 56; ++k) h[k] = fmaf(vj, sW0[j * 56 + k], h[k]);
  }
  const float iv = invc[n];
  const float* sr = sacc + (size_t)n * 16;
  for (int j = 0; j < 16; ++j) {
    const float vj = sr[j] * iv;
    #pragma unroll
    for (int k = 0; k < 56; ++k) h[k] = fmaf(vj, sW0[(64 + j) * 56 + k], h[k]);
  }
  float o[32];
  #pragma unroll
  for (int m = 0; m < 32; ++m) o[m] = sb1[m];
  for (int k = 0; k < 56; ++k) {
    const float hk = fmaxf(h[k], 0.f);
    #pragma unroll
    for (int m = 0; m < 32; ++m) o[m] = fmaf(hk, sW1[k * 32 + m], o[m]);
  }
  _Float16* nwr = nc + (size_t)n * 64 + 32;
  #pragma unroll
  for (int m = 0; m < 32; ++m) nwr[m] = (_Float16)fmaxf(o[m], 0.f);
}

extern "C" void kernel_launch(void* const* d_in, const int* in_sizes, int n_in,
                              void* d_out, int out_size, void* d_ws, size_t ws_size,
                              hipStream_t stream) {
  (void)in_sizes; (void)n_in; (void)out_size; (void)ws_size;
  const float* x         = (const float*)d_in[0];
  const int*   edge_idx  = (const int*)d_in[1];
  const float* edge_attr = (const float*)d_in[2];
  const float* enc_W0 = (const float*)d_in[3];
  const float* enc_b0 = (const float*)d_in[4];
  const float* enc_W1 = (const float*)d_in[5];
  const float* enc_b1 = (const float*)d_in[6];
  const float* enc_W2 = (const float*)d_in[7];
  const float* enc_b2 = (const float*)d_in[8];
  const float* edge_W0 = (const float*)d_in[9];
  const float* edge_b0 = (const float*)d_in[10];
  const float* edge_W1 = (const float*)d_in[11];
  const float* edge_b1 = (const float*)d_in[12];
  const float* node_W0 = (const float*)d_in[13];
  const float* node_b0 = (const float*)d_in[14];
  const float* node_W1 = (const float*)d_in[15];
  const float* node_b1 = (const float*)d_in[16];
  const int* rowp = edge_idx;
  const int* colp = edge_idx + N_EDGES;

  char* p = (char*)d_ws;
  auto walloc = [&](size_t bytes) -> void* {
    void* r = (void*)p;
    p += (bytes + 255) & ~(size_t)255;
    return r;
  };
  _Float16* w0f  = (_Float16*)walloc(25 * 512 * sizeof(_Float16));
  _Float16* w1f  = (_Float16*)walloc(3 * 512 * sizeof(_Float16));
  _Float16* nc   = (_Float16*)walloc((size_t)N_NODES * 64 * sizeof(_Float16));
  _Float16* e0b  = (_Float16*)walloc((size_t)N_EDGES * 16 * sizeof(_Float16));
  _Float16* ebA  = (_Float16*)walloc((size_t)N_EDGES * 16 * sizeof(_Float16));
  _Float16* ebB  = (_Float16*)walloc((size_t)N_EDGES * 16 * sizeof(_Float16));
  float*    sacc = (float*)walloc((size_t)N_NODES * 16 * sizeof(float));
  float*    invc = (float*)walloc((size_t)N_NODES * sizeof(float));

  prep_weights_kernel<<<56, 256, 0, stream>>>(edge_W0, edge_W1, w0f, w1f);
  node_init_kernel<<<(N_NODES + 255) / 256, 256, 0, stream>>>(x, nc);
  encoder_kernel<<<N_EDGES / 256, 256, 0, stream>>>(edge_attr, enc_W0, enc_b0,
                                                    enc_W1, enc_b1, enc_W2, enc_b2, e0b);
  // mean denominators are step-invariant: compute once
  zero_f32_kernel<<<(N_NODES + 255) / 256, 256, 0, stream>>>(invc, N_NODES);
  count_kernel<<<N_EDGES / 256, 256, 0, stream>>>(rowp, invc);
  inv_kernel<<<(N_NODES + 255) / 256, 256, 0, stream>>>(invc);

  _Float16* eprev = e0b;
  _Float16* bufs[2] = { ebA, ebB };
  for (int step = 1; step <= 6; ++step) {
    zero_f32_kernel<<<(N_NODES * 16 + 255) / 256, 256, 0, stream>>>(sacc, N_NODES * 16);
    _Float16* ecur = bufs[step & 1];
    float* outp = (step >= 2) ? ((float*)d_out + (size_t)(step - 2) * N_EDGES * 16)
                              : nullptr;
    edge_step_kernel<<<N_EDGES / 64, 128, 0, stream>>>(nc, e0b, eprev, rowp, colp,
                                                       w0f, w1f, edge_b0, edge_b1,
                                                       ecur, outp, sacc);
    if (step < 6)
      node_step_kernel<<<(N_NODES + 255) / 256, 256, 0, stream>>>(nc, sacc, invc,
                                                                  node_W0, node_b0,
                                                                  node_W1, node_b1);
    eprev = ecur;
  }
}
